// RadianceNetwork_32753420599335
// MI455X (gfx1250) — matrix operand order repaired and checked by reading, not compiled, not run on hardware
//
#include <hip/hip_runtime.h>
#include <hip/hip_bf16.h>
#include <math.h>

typedef _Float16 half_t;
typedef __attribute__((ext_vector_type(16))) _Float16 v16h;
typedef __attribute__((ext_vector_type(8)))  _Float16 v8h;
typedef __attribute__((ext_vector_type(8)))  float    v8f;

// ---------------- workspace layout (units: halves) ----------------
constexpr int OFF_W1_0 = 0;                    // 256 x 32   (3 -> pad 32)
constexpr int OFF_W1_1 = OFF_W1_0 + 256 * 32;  // 256 x 256
constexpr int OFF_W1_2 = OFF_W1_1 + 256 * 256;
constexpr int OFF_W1_3 = OFF_W1_2 + 256 * 256;
constexpr int OFF_W2_0 = OFF_W1_3 + 256 * 256; // 256 x 288  (259 -> pad 288)
constexpr int OFF_W2_1 = OFF_W2_0 + 256 * 288;
constexpr int OFF_W2_2 = OFF_W2_1 + 256 * 256;
constexpr int OFF_W2_3 = OFF_W2_2 + 256 * 256; // 272 x 256  (257 rows -> pad 272)
constexpr int OFF_C0   = OFF_W2_3 + 272 * 256; // 128 x 288
constexpr int OFF_C1   = OFF_C0   + 128 * 288; // 16 x 128   (3 rows -> pad 16)

// Strides of 296 halves = 148 dwords; 148*r mod 64 distinct for r=0..15
// -> conflict-free row-parallel LDS reads for A rows and B weight rows.
constexpr int AST  = 296;                  // activation row stride (halves)
constexpr int WST  = 296;                  // LDS weight row stride (halves)
constexpr int BUFH = 16 * AST;             // per-wave activation slab (halves)
constexpr int ACT_HALVES = 8 * BUFH;       // 8 waves
constexpr int HROWS = 144;                 // max weight rows per half-phase
constexpr int HBUF  = HROWS * WST;         // half-phase weight buffer (halves)
constexpr int LDS_BYTES = (ACT_HALVES + 2 * HBUF) * 2;  // 246,272 B < 320 KB

// ---------------- weight convert + pad (fp32 -> f16) ----------------
__global__ void cvt_pad_f16(const float* __restrict__ src, half_t* __restrict__ dst,
                            int sR, int sC, int dR, int dC) {
  int i = blockIdx.x * blockDim.x + threadIdx.x;
  if (i >= dR * dC) return;
  int r = i / dC, c = i - r * dC;
  float v = (r < sR && c < sC) ? src[r * sC + c] : 0.0f;
  dst[i] = (half_t)v;
}

// ---------------- fragment loads ----------------
// A (16x32 f16): lane<16 holds K k0+0..7 / k0+16..23, lane>=16 holds
// K k0+8..15 / k0+24..31, row = lane&15. Source: this wave's LDS activations.
__device__ __forceinline__ v16h load_a_frag(const half_t* act, int k0, int lane) {
  int row = lane & 15;
  int kk  = k0 + ((lane >> 4) << 3);
  const half_t* p = act + row * AST + kk;
  v8h lo = *(const v8h*)(p);
  v8h hi = *(const v8h*)(p + 16);
  return __builtin_shufflevector(lo, hi, 0, 1, 2, 3, 4, 5, 6, 7,
                                         8, 9, 10, 11, 12, 13, 14, 15);
}

// B (32x16 f16): lane<16 -> n=lane, K k0+0..7 / k0+16..23; lane>=16 -> n=lane-16,
// K k0+8..15 / k0+24..31. Source: tile base (16 rows) inside LDS half-phase buffer.
__device__ __forceinline__ v16h load_b_frag(const half_t* wt, int k0, int lane) {
  int n  = lane & 15;
  int kk = k0 + ((lane >> 4) << 3);
  const half_t* p = wt + n * WST + kk;
  v8h lo = *(const v8h*)(p);        // 16B aligned (WST*2 = 592 = 37*16)
  v8h hi = *(const v8h*)(p + 16);
  return __builtin_shufflevector(lo, hi, 0, 1, 2, 3, 4, 5, 6, 7,
                                         8, 9, 10, 11, 12, 13, 14, 15);
}

__device__ __forceinline__ v8f wmma_f16(v16h a, v16h b, v8f c) {
  return __builtin_amdgcn_wmma_f32_16x16x32_f16(false, a, false, b,
                                                (short)0, c, false, false);
}

// ---------------- async weight staging (CDNA5 ASYNC path) ----------------
// Fire-and-forget copy of `rows` weight rows (row stride K halves) starting at
// global row g0 into LDS half-phase buffer `bufIdx`. Tracked by ASYNCcnt; no
// VGPR round trip, overlaps with the WMMA stream of the current phase.
__device__ __forceinline__ void async_stage(int bufIdx,
                                            const half_t* __restrict__ W,
                                            int g0, int rows, int K, int tid) {
  const int wave = tid >> 5, lane = tid & 31;
  const unsigned base = (unsigned)(ACT_HALVES + bufIdx * HBUF) * 2u;
  for (int r = wave; r < rows; r += 8) {
    const half_t* src = W + (size_t)(g0 + r) * K;
    const unsigned dstRow = base + (unsigned)r * (WST * 2);
    for (int c = lane * 8; c < K; c += 32 * 8) {
      unsigned dst = dstRow + (unsigned)c * 2u;
      asm volatile("global_load_async_to_lds_b128 %0, %1, off"
                   :: "v"(dst), "v"(src + c)
                   : "memory");
    }
  }
}

__device__ __forceinline__ void async_wait_bar() {
  asm volatile("s_wait_asynccnt 0x0" ::: "memory");  // own async fills done
  __syncthreads();                                   // everyone's fills done
}

// ---------------- compute one half-phase of a layer (in place) ----------------
// A slab lives in registers, so the wave's activation buffer is overwritten
// with this layer's output columns as tiles complete.
template <int K, bool GAUSS>
__device__ __forceinline__ void compute_phase(const v16h* afrag, half_t* act,
                                              const half_t* wbuf, int ntBegin,
                                              int ntEnd,
                                              const float* __restrict__ bias,
                                              const float* __restrict__ scale,
                                              int lane) {
  constexpr int NK = K / 32;
  const int nlane = lane & 15;
  const int rbase = (lane >> 4) << 3;
  for (int nt = ntBegin; nt < ntEnd; ++nt) {
    const half_t* wt = wbuf + (nt - ntBegin) * (16 * WST);
    v8f acc = {};
#pragma unroll
    for (int kt = 0; kt < NK; ++kt)
      acc = wmma_f16(afrag[kt], load_b_frag(wt, kt * 32, lane), acc);
    const int n  = nt * 16 + nlane;
    const float bn = bias[n];
    const float sn = GAUSS ? scale[n] : 0.0f;
#pragma unroll
    for (int r = 0; r < 8; ++r) {
      float x = acc[r] + bn;
      if (GAUSS) { float t = x * sn; x = __expf(-0.5f * t * t); }
      act[(rbase + r) * AST + n] = (half_t)x;
    }
  }
}

template <int K>
__device__ __forceinline__ void load_a_all(v16h* afrag, const half_t* act,
                                           int lane) {
#pragma unroll
  for (int kt = 0; kt < K / 32; ++kt) afrag[kt] = load_a_frag(act, kt * 32, lane);
}

// stage 3-wide global vectors (pos/dir) into 32 padded LDS columns at colBase
__device__ __forceinline__ void stage_vec3(half_t* buf, const float* __restrict__ g,
                                           int rowbase, int colBase, int lane) {
  for (int i = lane; i < 16 * 32; i += 32) {
    int r = i >> 5, c = i & 31;
    float v = (c < 3) ? g[(size_t)(rowbase + r) * 3 + c] : 0.0f;
    buf[r * AST + colBase + c] = (half_t)v;
  }
}

// ---------------- fused radiance network ----------------
__global__ void __launch_bounds__(256, 1)
radiance_fused(const float* __restrict__ pos, const float* __restrict__ dir,
               const half_t* __restrict__ W,
               const float* __restrict__ b10, const float* __restrict__ b11,
               const float* __restrict__ b12, const float* __restrict__ b13,
               const float* __restrict__ s10, const float* __restrict__ s11,
               const float* __restrict__ s12, const float* __restrict__ s13,
               const float* __restrict__ b20, const float* __restrict__ b21,
               const float* __restrict__ b22, const float* __restrict__ b23,
               const float* __restrict__ s20, const float* __restrict__ s21,
               const float* __restrict__ s22,
               const float* __restrict__ cb0, const float* __restrict__ cb1,
               const float* __restrict__ cs,
               float* __restrict__ out, int n) {
  extern __shared__ char smem_raw[];
  half_t* smem = (half_t*)smem_raw;

  const int tid  = threadIdx.x;
  const int lane = tid & 31;
  const int wave = tid >> 5;
  const int rowbase = blockIdx.x * 128 + wave * 16;
  if (blockIdx.x * 128 >= n) return;

  half_t* act = smem + wave * BUFH;       // wave-private activation slab
  half_t* wb0 = smem + ACT_HALVES;        // half-phase weight buffers
  half_t* wb1 = wb0 + HBUF;
  const int nlane = lane & 15;
  const int rbase = (lane >> 4) << 3;

  v16h afrag[9];

  // ---- prologue: input + first half-phase of first layer ----
  stage_vec3(act, pos, rowbase, 0, lane);      // act[:,0:32] = [pos | 0]
  async_stage(0, W + OFF_W1_0, 0, 128, 32, tid);
  async_wait_bar();

  // ---- model_density_1: 4x (Linear + GaussAct) ----
  load_a_all<32>(afrag, act, lane);
  async_stage(1, W + OFF_W1_0, 128, 128, 32, tid);
  compute_phase<32, true>(afrag, act, wb0, 0, 8, b10, s10, lane);
  async_wait_bar();
  async_stage(0, W + OFF_W1_1, 0, 128, 256, tid);
  compute_phase<32, true>(afrag, act, wb1, 8, 16, b10, s10, lane);
  async_wait_bar();

  load_a_all<256>(afrag, act, lane);
  async_stage(1, W + OFF_W1_1, 128, 128, 256, tid);
  compute_phase<256, true>(afrag, act, wb0, 0, 8, b11, s11, lane);
  async_wait_bar();
  async_stage(0, W + OFF_W1_2, 0, 128, 256, tid);
  compute_phase<256, true>(afrag, act, wb1, 8, 16, b11, s11, lane);
  async_wait_bar();

  load_a_all<256>(afrag, act, lane);
  async_stage(1, W + OFF_W1_2, 128, 128, 256, tid);
  compute_phase<256, true>(afrag, act, wb0, 0, 8, b12, s12, lane);
  async_wait_bar();
  async_stage(0, W + OFF_W1_3, 0, 128, 256, tid);
  compute_phase<256, true>(afrag, act, wb1, 8, 16, b12, s12, lane);
  async_wait_bar();

  load_a_all<256>(afrag, act, lane);
  async_stage(1, W + OFF_W1_3, 128, 128, 256, tid);
  compute_phase<256, true>(afrag, act, wb0, 0, 8, b13, s13, lane);
  async_wait_bar();
  async_stage(0, W + OFF_W2_0, 0, 128, 288, tid);
  compute_phase<256, true>(afrag, act, wb1, 8, 16, b13, s13, lane);
  async_wait_bar();

  // ---- concat(z, pos): act[:, 256:288] = [pos | 0] ----
  stage_vec3(act, pos, rowbase, 256, lane);

  // ---- model_density_2 ----
  load_a_all<288>(afrag, act, lane);
  async_stage(1, W + OFF_W2_0, 128, 128, 288, tid);
  compute_phase<288, true>(afrag, act, wb0, 0, 8, b20, s20, lane);
  async_wait_bar();
  async_stage(0, W + OFF_W2_1, 0, 128, 256, tid);
  compute_phase<288, true>(afrag, act, wb1, 8, 16, b20, s20, lane);
  async_wait_bar();

  load_a_all<256>(afrag, act, lane);
  async_stage(1, W + OFF_W2_1, 128, 128, 256, tid);
  compute_phase<256, true>(afrag, act, wb0, 0, 8, b21, s21, lane);
  async_wait_bar();
  async_stage(0, W + OFF_W2_2, 0, 128, 256, tid);
  compute_phase<256, true>(afrag, act, wb1, 8, 16, b21, s21, lane);
  async_wait_bar();

  load_a_all<256>(afrag, act, lane);
  async_stage(1, W + OFF_W2_2, 128, 128, 256, tid);
  compute_phase<256, true>(afrag, act, wb0, 0, 8, b22, s22, lane);
  async_wait_bar();
  async_stage(0, W + OFF_W2_3, 0, 128, 256, tid);
  compute_phase<256, true>(afrag, act, wb1, 8, 16, b22, s22, lane);
  async_wait_bar();

  // ---- final d2 layer (272x256 padded): features + density column ----
  load_a_all<256>(afrag, act, lane);
  async_stage(1, W + OFF_W2_3, 128, 144, 256, tid);  // tiles 8..16 (144 rows)
  compute_phase<256, false>(afrag, act, wb0, 0, 8, b23, nullptr, lane);
  async_wait_bar();
  async_stage(0, W + OFF_C0, 0, 128, 288, tid);
  compute_phase<256, false>(afrag, act, wb1, 8, 16, b23, nullptr, lane);
  {
    // density tile: global rows 256..271 -> local rows 128..143 of wb1
    const half_t* wt = wb1 + 8 * (16 * WST);
    v8f acc = {};
#pragma unroll
    for (int kt = 0; kt < 8; ++kt)
      acc = wmma_f16(afrag[kt], load_b_frag(wt, kt * 32, lane), acc);
    if (nlane == 0) {
      const float bb = b23[256];
#pragma unroll
      for (int r = 0; r < 8; ++r) {
        float x = acc[r] + bb - 1.0f;
        float sp = (x > 20.0f) ? x : log1pf(__expf(x));
        out[(size_t)3 * n + rowbase + rbase + r] = sp;   // density block
      }
    }
  }
  async_wait_bar();

  // ---- concat(features, dir): act[:, 256:288] = [dir | 0] ----
  stage_vec3(act, dir, rowbase, 256, lane);

  // ---- model_color ----
  load_a_all<288>(afrag, act, lane);
  async_stage(1, W + OFF_C1, 0, 16, 128, tid);
  compute_phase<288, true>(afrag, act, wb0, 0, 8, cb0, cs, lane);
  async_wait_bar();
  {
    load_a_all<128>(afrag, act, lane);
    v8f acc = {};
#pragma unroll
    for (int kt = 0; kt < 4; ++kt)
      acc = wmma_f16(afrag[kt], load_b_frag(wb1, kt * 32, lane), acc);
    if (nlane < 3) {
      const float bb = cb1[nlane];
#pragma unroll
      for (int r = 0; r < 8; ++r) {
        float x = acc[r] + bb;
        float sg = 1.0f / (1.0f + __expf(-x));
        out[(size_t)(rowbase + rbase + r) * 3 + nlane] = sg;  // rgb block
      }
    }
  }
}

// ---------------- host launch ----------------
static inline void cvt(const void* src, half_t* dst, int sR, int sC, int dR, int dC,
                       hipStream_t stream) {
  int total = dR * dC;
  cvt_pad_f16<<<(total + 255) / 256, 256, 0, stream>>>((const float*)src, dst,
                                                       sR, sC, dR, dC);
}

extern "C" void kernel_launch(void* const* d_in, const int* in_sizes, int n_in,
                              void* d_out, int out_size, void* d_ws, size_t ws_size,
                              hipStream_t stream) {
  const float* pos = (const float*)d_in[0];
  const float* dir = (const float*)d_in[1];
  const float* d1w[4] = {(const float*)d_in[2], (const float*)d_in[3],
                         (const float*)d_in[4], (const float*)d_in[5]};
  const float* d1b[4] = {(const float*)d_in[6], (const float*)d_in[7],
                         (const float*)d_in[8], (const float*)d_in[9]};
  const float* d1s[4] = {(const float*)d_in[10], (const float*)d_in[11],
                         (const float*)d_in[12], (const float*)d_in[13]};
  const float* d2w[4] = {(const float*)d_in[14], (const float*)d_in[15],
                         (const float*)d_in[16], (const float*)d_in[17]};
  const float* d2b[4] = {(const float*)d_in[18], (const float*)d_in[19],
                         (const float*)d_in[20], (const float*)d_in[21]};
  const float* d2s[3] = {(const float*)d_in[22], (const float*)d_in[23],
                         (const float*)d_in[24]};
  const float* cw0 = (const float*)d_in[25];
  const float* cw1 = (const float*)d_in[26];
  const float* cb0 = (const float*)d_in[27];
  const float* cb1 = (const float*)d_in[28];
  const float* cs  = (const float*)d_in[29];

  half_t* W = (half_t*)d_ws;

  // f16 weight staging (~1 MB total, stays resident in 192 MB L2)
  cvt(d1w[0], W + OFF_W1_0, 256, 3,   256, 32,  stream);
  cvt(d1w[1], W + OFF_W1_1, 256, 256, 256, 256, stream);
  cvt(d1w[2], W + OFF_W1_2, 256, 256, 256, 256, stream);
  cvt(d1w[3], W + OFF_W1_3, 256, 256, 256, 256, stream);
  cvt(d2w[0], W + OFF_W2_0, 256, 259, 256, 288, stream);
  cvt(d2w[1], W + OFF_W2_1, 256, 256, 256, 256, stream);
  cvt(d2w[2], W + OFF_W2_2, 256, 256, 256, 256, stream);
  cvt(d2w[3], W + OFF_W2_3, 257, 256, 272, 256, stream);
  cvt(cw0,    W + OFF_C0,   128, 259, 128, 288, stream);
  cvt(cw1,    W + OFF_C1,   3,   128, 16,  128, stream);

  const int n = in_sizes[0] / 3;         // number of samples
  const int blocks = (n + 127) / 128;    // 128 rows per block (8 waves x 16)
  radiance_fused<<<blocks, 256, LDS_BYTES, stream>>>(
      pos, dir, W,
      d1b[0], d1b[1], d1b[2], d1b[3],
      d1s[0], d1s[1], d1s[2], d1s[3],
      d2b[0], d2b[1], d2b[2], d2b[3],
      d2s[0], d2s[1], d2s[2],
      cb0, cb1, cs,
      (float*)d_out, n);
}
